// Model_45389214384852
// MI455X (gfx1250) — matrix operand order repaired and checked by reading, not compiled, not run on hardware
//
#include <hip/hip_runtime.h>
#include <hip/hip_bf16.h>

typedef _Float16 f16;
typedef _Float16 v16h __attribute__((ext_vector_type(16)));
typedef float v8f __attribute__((ext_vector_type(8)));

__device__ __forceinline__ v8f v8f_zero() {
  v8f z = {0.f, 0.f, 0.f, 0.f, 0.f, 0.f, 0.f, 0.f};
  return z;
}

// Build a 16-half WMMA fragment from two 16-byte LDS chunks.
__device__ __forceinline__ v16h load_frag(const f16* p0, const f16* p1) {
  union { v16h h; uint4 u[2]; } f;
  f.u[0] = *(const uint4*)p0;
  f.u[1] = *(const uint4*)p1;
  return f.h;
}

// CDNA5 async global->LDS copy, 16 bytes per lane (ASYNCcnt tracked).
// LDS byte address = low 32 bits of the flat shared pointer.
__device__ __forceinline__ void async_copy_b128(void* lds_dst, const void* gsrc) {
  uint32_t loff = (uint32_t)(uintptr_t)lds_dst;
  asm volatile("global_load_async_to_lds_b128 %0, %1, off"
               :: "v"(loff), "v"(gsrc)
               : "memory");
}
__device__ __forceinline__ void wait_async0() {
  asm volatile("s_wait_asynccnt 0x0" ::: "memory");
}

// ---------------------------------------------------------------------------
// LayerNorm: one wave per row; writes f16 normalized row padded with zeros.
// ---------------------------------------------------------------------------
__global__ __launch_bounds__(128) void ln_kernel(
    const float* __restrict__ x, const float* __restrict__ g,
    const float* __restrict__ b, f16* __restrict__ xn, int hid, int Kp) {
  int row = blockIdx.x * 4 + (threadIdx.x >> 5);
  int lane = threadIdx.x & 31;
  const float* xr = x + (size_t)row * hid;
  float v[6];
  int cnt = 0;
  float s = 0.f;
  for (int i = lane; i < hid; i += 32) { float t = xr[i]; v[cnt++] = t; s += t; }
#pragma unroll
  for (int off = 16; off; off >>= 1) s += __shfl_xor(s, off, 32);
  float mu = s / (float)hid;
  float s2 = 0.f;
  for (int j = 0; j < cnt; ++j) { float d = v[j] - mu; s2 += d * d; }
#pragma unroll
  for (int off = 16; off; off >>= 1) s2 += __shfl_xor(s2, off, 32);
  float rstd = rsqrtf(s2 / (float)hid + 1e-12f);
  f16* o = xn + (size_t)row * Kp;
  int j = 0;
  for (int i = lane; i < hid; i += 32) o[i] = (f16)((v[j++] - mu) * rstd * g[i] + b[i]);
  for (int i = lane; i < Kp; i += 32)
    if (i >= hid) o[i] = (f16)0.f;
}

// ---------------------------------------------------------------------------
// Tiled WMMA GEMM: C[64x64 per WG] = A(f16, lda=Kp) * W(f32 KxN -> f16).
// A tile staged via async global->LDS; W tile converted f32->f16 on VALU path.
// ---------------------------------------------------------------------------
template <int F16OUT>
__global__ __launch_bounds__(128) void gemm_wmma(
    const f16* __restrict__ A, int lda, const float* __restrict__ W, int Kv,
    int N, int Ksteps, void* __restrict__ Cout, int ldc, float scale) {
  __shared__ __align__(16) f16 sA[64][32];    // A tile, row-major
  __shared__ __align__(16) f16 sBt[64][32];   // W tile transposed: [n][k]
  int row0 = blockIdx.y * 64;
  int col0 = blockIdx.x * 64;
  int tid = threadIdx.x;
  int w = tid >> 5, L = tid & 31;
  int lane16 = L & 15;
  int hi = L >> 4;
  v8f acc[4];
#pragma unroll
  for (int i = 0; i < 4; ++i) acc[i] = v8f_zero();

  for (int ks = 0; ks < Ksteps; ++ks) {
    __syncthreads();  // protect LDS reuse from previous iteration's readers
#pragma unroll 2
    for (int i = tid; i < 256; i += 128) {
      int r = i >> 2, seg = i & 3;
      async_copy_b128(&sA[r][seg * 8],
                      A + (size_t)(row0 + r) * lda + ks * 32 + seg * 8);
    }
    for (int i = tid; i < 2048; i += 128) {
      int k = i >> 6, nn = i & 63;
      int gk = ks * 32 + k, gn = col0 + nn;
      float wv = (gk < Kv && gn < N) ? W[(size_t)gk * N + gn] : 0.f;
      sBt[nn][k] = (f16)wv;
    }
    wait_async0();
    __syncthreads();
    int ar = 16 * w + lane16;
    int kb = hi * 8;
    int kseg = hi * 16;
    v16h af = load_frag(&sA[ar][kb], &sA[ar][16 + kb]);
#pragma unroll
    for (int nt = 0; nt < 4; ++nt) {
      int bn = nt * 16 + lane16;
      v16h bf = load_frag(&sBt[bn][kb], &sBt[bn][16 + kb]);
      acc[nt] = __builtin_amdgcn_wmma_f32_16x16x32_f16(
          false, af, false, bf, (short)0, acc[nt], false, false);
    }
  }
  int mrow = row0 + 16 * w + hi * 8;
  int colb = col0 + lane16;
#pragma unroll
  for (int nt = 0; nt < 4; ++nt) {
    int gc = colb + nt * 16;
    if (gc < N) {
#pragma unroll
      for (int r = 0; r < 8; ++r) {
        float val = acc[nt][r] * scale;
        if (F16OUT)
          ((f16*)Cout)[(size_t)(mrow + r) * ldc + gc] = (f16)val;
        else
          ((float*)Cout)[(size_t)(mrow + r) * ldc + gc] = val;
      }
    }
  }
}

// ---------------------------------------------------------------------------
// Chunked local attention: one WG (4 waves) per (b, head, chunk).
// Q/K tiles staged via async global->LDS, QK^T via WMMA (d=16 zero-padded to
// K=32), register softmax (wave32 shuffles), P staged through LDS for the
// operand transpose, PV via WMMA.
// ---------------------------------------------------------------------------
__global__ __launch_bounds__(128) void attn_wmma(
    const f16* __restrict__ Q, const f16* __restrict__ K,
    const f16* __restrict__ V, f16* __restrict__ ctx,
    const int* __restrict__ mask, int S, int H, int HD, int nch) {
  __shared__ __align__(16) f16 sQ[64][32];    // [q][d padded to 32]
  __shared__ __align__(16) f16 sK[128][32];   // [key][d padded to 32]
  __shared__ __align__(16) f16 sVt[16][128];  // V^T: [d][key]
  __shared__ __align__(16) f16 sP[64][128];   // probs, f16
  __shared__ int sM[128];

  int bid = blockIdx.x;
  int c = bid % nch;
  int th = bid / nch;
  int h = th % H;
  int b = th / H;
  int tok0 = b * S + c * 64;
  int tokp = b * S + ((c + nch - 1) % nch) * 64;  // look-back chunk (wraps)
  int tid = threadIdx.x;
  uint4 z4 = make_uint4(0u, 0u, 0u, 0u);

  {  // Q tile: 64 rows x 16 halves (two b128 segs per row), async staged
    int r = tid >> 1, seg = tid & 1;
    async_copy_b128(&sQ[r][seg * 8],
                    Q + (size_t)(tok0 + r) * HD + h * 16 + seg * 8);
    *(uint4*)&sQ[r][16 + seg * 8] = z4;
  }
#pragma unroll 2
  for (int i = tid; i < 256; i += 128) {  // K tiles: prev chunk then current
    int r = i >> 1, seg = i & 1;
    int tok = (r < 64) ? (tokp + r) : (tok0 + r - 64);
    async_copy_b128(&sK[r][seg * 8],
                    K + (size_t)tok * HD + h * 16 + seg * 8);
    *(uint4*)&sK[r][16 + seg * 8] = z4;
  }
  for (int i = tid; i < 2048; i += 128) {  // V^T (transpose on VALU path)
    int key = i >> 4, d = i & 15;
    int tok = (key < 64) ? (tokp + key) : (tok0 + key - 64);
    sVt[d][key] = V[(size_t)tok * HD + h * 16 + d];
  }
  {
    int key = tid;
    int tok = (key < 64) ? (tokp + key) : (tok0 + key - 64);
    sM[key] = mask ? mask[tok] : 1;
  }
  wait_async0();
  __syncthreads();

  int w = tid >> 5, L = tid & 31;
  int lane16 = L & 15;
  int hi = L >> 4;
  int kb = hi * 8;
  int kseg = hi * 16;
  int ar = 16 * w + lane16;

  // scores for this wave's 16 query rows, all 128 keys (8 N-tiles)
  v16h qf = load_frag(&sQ[ar][kb], &sQ[ar][16 + kb]);
  v8f sc[8];
#pragma unroll
  for (int nt = 0; nt < 8; ++nt) {
    v16h kf =
        load_frag(&sK[nt * 16 + lane16][kb], &sK[nt * 16 + lane16][16 + kb]);
    sc[nt] = __builtin_amdgcn_wmma_f32_16x16x32_f16(
        false, qf, false, kf, (short)0, v8f_zero(), false, false);
  }

  // masked softmax across 128 keys: row r lives on 16 lanes of this half-wave
  float rmax[8], rsum[8];
#pragma unroll
  for (int r = 0; r < 8; ++r) rmax[r] = -3.0e38f;
#pragma unroll
  for (int nt = 0; nt < 8; ++nt) {
    bool ok = sM[nt * 16 + lane16] != 0;
#pragma unroll
    for (int r = 0; r < 8; ++r) {
      float vv = sc[nt][r];
      if (!ok) vv = -1e9f;  // HF replaces with MASK_VAL
      sc[nt][r] = vv;
      rmax[r] = fmaxf(rmax[r], vv);
    }
  }
#pragma unroll
  for (int off = 1; off < 16; off <<= 1)
#pragma unroll
    for (int r = 0; r < 8; ++r)
      rmax[r] = fmaxf(rmax[r], __shfl_xor(rmax[r], off, 32));
#pragma unroll
  for (int r = 0; r < 8; ++r) rsum[r] = 0.f;
#pragma unroll
  for (int nt = 0; nt < 8; ++nt)
#pragma unroll
    for (int r = 0; r < 8; ++r) {
      float e = __expf(sc[nt][r] - rmax[r]);
      sc[nt][r] = e;
      rsum[r] += e;
    }
#pragma unroll
  for (int off = 1; off < 16; off <<= 1)
#pragma unroll
    for (int r = 0; r < 8; ++r) rsum[r] += __shfl_xor(rsum[r], off, 32);

  int prow0 = 16 * w + hi * 8;
#pragma unroll
  for (int nt = 0; nt < 8; ++nt)
#pragma unroll
    for (int r = 0; r < 8; ++r)
      sP[prow0 + r][nt * 16 + lane16] = (f16)(sc[nt][r] / rsum[r]);
  __syncthreads();

  // out = P @ Vcat : contraction over 128 keys in 4 WMMA k-steps
  v8f acc = v8f_zero();
#pragma unroll
  for (int kb2 = 0; kb2 < 4; ++kb2) {
    v16h pf = load_frag(&sP[ar][kb2 * 32 + kb], &sP[ar][kb2 * 32 + 16 + kb]);
    v16h vf = load_frag(&sVt[lane16][kb2 * 32 + kb],
                        &sVt[lane16][kb2 * 32 + 16 + kb]);
    acc = __builtin_amdgcn_wmma_f32_16x16x32_f16(false, pf, false, vf, (short)0,
                                                 acc, false, false);
  }
  int orow = tok0 + prow0;
#pragma unroll
  for (int r = 0; r < 8; ++r)
    ctx[(size_t)(orow + r) * HD + h * 16 + lane16] = (f16)acc[r];
}

// ---------------------------------------------------------------------------
extern "C" void kernel_launch(void* const* d_in, const int* in_sizes, int n_in,
                              void* d_out, int out_size, void* d_ws,
                              size_t ws_size, hipStream_t stream) {
  (void)in_sizes; (void)n_in; (void)out_size; (void)ws_size;
  const float* x = (const float*)d_in[0];
  const float* y = (const float*)d_in[1];
  const int* mask = (const int*)d_in[2];
  const float* ln0_g = (const float*)d_in[3];
  const float* ln0_b = (const float*)d_in[4];
  const float* wq0 = (const float*)d_in[5];
  const float* wk0 = (const float*)d_in[6];
  const float* wv0 = (const float*)d_in[7];
  const float* wo0 = (const float*)d_in[8];
  const float* ln1_g = (const float*)d_in[9];
  const float* ln1_b = (const float*)d_in[10];
  const float* wq1 = (const float*)d_in[11];
  const float* wk1 = (const float*)d_in[12];
  const float* wv1 = (const float*)d_in[13];
  const float* wo1 = (const float*)d_in[14];

  const int Bz = 4, S = 8192;
  const int R = Bz * S;     // 32768 token rows
  const int nch = S / 64;   // 128 chunks

  // Scratch (f16), sized for the larger block; blocks run serially on stream.
  char* ws = (char*)d_ws;
  size_t seg = (size_t)R * 192 * sizeof(f16);
  f16* xn = (f16*)(ws + 0 * seg);
  f16* qb = (f16*)(ws + 1 * seg);
  f16* kb = (f16*)(ws + 2 * seg);
  f16* vb = (f16*)(ws + 3 * seg);
  f16* cx = (f16*)(ws + 4 * seg);

  auto run = [&](const float* inp, int hid, int Kp, int H, const float* g,
                 const float* bb, const float* wq, const float* wk,
                 const float* wv, const float* wo, const int* m, float* outp) {
    int HD = H * 16;
    ln_kernel<<<R / 4, 128, 0, stream>>>(inp, g, bb, xn, hid, Kp);
    dim3 gq((HD + 63) / 64, R / 64);
    gemm_wmma<1><<<gq, 128, 0, stream>>>(xn, Kp, wq, hid, HD, Kp / 32, qb, HD, 1.0f);
    gemm_wmma<1><<<gq, 128, 0, stream>>>(xn, Kp, wk, hid, HD, Kp / 32, kb, HD, 0.25f);
    gemm_wmma<1><<<gq, 128, 0, stream>>>(xn, Kp, wv, hid, HD, Kp / 32, vb, HD, 1.0f);
    attn_wmma<<<Bz * H * nch, 128, 0, stream>>>(qb, kb, vb, cx, m, S, H, HD, nch);
    dim3 go((hid + 63) / 64, R / 64);
    gemm_wmma<0><<<go, 128, 0, stream>>>(cx, HD, wo, HD, hid, HD / 32, outp, hid, 1.0f);
  };

  float* out0 = (float*)d_out;
  float* out1 = out0 + (size_t)R * 192;
  run(x, 192, 192, 12, ln0_g, ln0_b, wq0, wk0, wv0, wo0, nullptr, out0);
  run(y, 66, 96, 6, ln1_g, ln1_b, wq1, wk1, wv1, wo1, mask, out1);
}
